// PlaneFormer_39625368273030
// MI455X (gfx1250) — matrix operand order repaired and checked by reading, not compiled, not run on hardware
//
#include <hip/hip_runtime.h>
#include <hip/hip_bf16.h>

typedef __attribute__((ext_vector_type(16))) __bf16 v16bf;
typedef __attribute__((ext_vector_type(8)))  float  v8f;

union FragBF { v16bf v; uint4 u[2]; };

// ---------------- workspace layout (bytes) ----------------
#define WS_EMBB 0x000000u   // 64*32*256 bf16 = 1 MB
#define WS_G1B  0x100000u   // 64*256 bf16 = 32 KB
#define WS_G2B  0x108000u   // 64*256 bf16 = 32 KB
#define WS_WPK  0x110000u   // packed bf16 weights, 2,789,376 elems = 5.58 MB
#define WS_CAM  0x662000u   // 65536 f32
#define WS_ROT  0x6A2000u   // 65536*4 f32
#define WS_TRN  0x7A2000u   // 65536*4 f32 (3 used)

// packed weight layer offsets (bf16 elements), per-layer all-heads contiguous
#define WOFF_L0 0u
#define WOFF_L1 2097152u
#define WOFF_L2 2621440u
#define WOFF_L3 2752512u
#define WOFF_L4 2785280u

// ---------------- prep kernels ----------------
__global__ void conv_emb_kernel(const float* __restrict__ emb, __bf16* __restrict__ embb, int n) {
    int i = blockIdx.x * 256 + threadIdx.x;
    if (i < n) embb[i] = (__bf16)emb[i];
}

__global__ void prep_g_kernel(const float* __restrict__ emb, const int* __restrict__ np,
                              __bf16* __restrict__ g1b, __bf16* __restrict__ g2b) {
    int b = blockIdx.x, d = threadIdx.x;       // 64 blocks, 256 threads
    int n0 = np[b * 2], n1 = np[b * 2 + 1];
    float s1 = 0.f, s2 = 0.f;
    for (int t = 0; t < 32; ++t) {
        float e = emb[(b * 32 + t) * 256 + d];
        if (t < n0)           s1 += e;
        else if (t < n0 + n1) s2 += e;
    }
    g1b[b * 256 + d] = (__bf16)(s1 / (float)n0);
    g2b[b * 256 + d] = (__bf16)(s2 / (float)n1);
}

// pack w (H,K,N) fp32 -> WMMA-B tile order bf16:
// idx = head*(Nt*Kt*512) + ((nt*Kt + kt)*512) + lane*16 + j16
// element: n = nt*16 + (lane&15); k = kt*32 + (j16>>3)*16 + (lane>>4)*8 + (j16&7)
__global__ void pack_w_kernel(const float* __restrict__ w, __bf16* __restrict__ dst,
                              int K, int N, int Kt, int Nt, int total) {
    int i = blockIdx.x * 256 + threadIdx.x;
    if (i >= total) return;
    int perHead = Nt * Kt * 512;
    int head = i / perHead;
    int r = i % perHead;
    int tile = r >> 9;
    int lane = (r >> 4) & 31;
    int j16 = r & 15;
    int nt = tile / Kt, kt = tile % Kt;
    int n = nt * 16 + (lane & 15);
    int k = kt * 32 + ((j16 >> 3) << 4) + ((lane >> 4) << 3) + (j16 & 7);
    float v = (n < N) ? w[(head * K + k) * N + n] : 0.f;
    dst[i] = (__bf16)v;
}

// ---------------- fused MLP kernel ----------------
// Generic LDS->LDS layer: A rows are wave-private, row-major bf16 in LDS.
__device__ __forceinline__ void layer_lds(const __bf16* __restrict__ wl, const float* __restrict__ bl,
                                          const char* inbuf, char* outbuf,
                                          int Kt, int Nt, int Nout, int Kin, int wave, int lane) {
    const int g = lane >> 4, ml = lane & 15;
    const char* arow = inbuf + (wave * 16 + ml) * Kin * 2;
    for (int nt = 0; nt < Nt; ++nt) {
        v8f c = {};
        const __bf16* wb = wl + nt * Kt * 512 + lane * 16;
        for (int kt = 0; kt < Kt; ++kt) {
            FragBF a, wf;
            const int off = (kt * 32 + (g << 3)) * 2;
            a.u[0]  = *(const uint4*)(arow + off);
            a.u[1]  = *(const uint4*)(arow + off + 32);
            wf.u[0] = *(const uint4*)(wb);
            wf.u[1] = *(const uint4*)(wb + 8);
            wb += 512;
            c = __builtin_amdgcn_wmma_f32_16x16x32_bf16(false, a.v, false, wf.v, (short)0, c, false, false);
        }
        const int nabs = nt * 16 + ml;
        const float bias = bl[nabs];
        const int mbase = wave * 16 + (g << 3);
#pragma unroll
        for (int v = 0; v < 8; ++v) {
            float x = c[v] + bias;
            x = x > 0.f ? x : 0.f;
            *(__bf16*)(outbuf + ((mbase + v) * Nout + nabs) * 2) = (__bf16)x;
        }
    }
}

__global__ void __launch_bounds__(256, 1) mlp_kernel(
    const __bf16* __restrict__ embb, const __bf16* __restrict__ g1b, const __bf16* __restrict__ g2b,
    const __bf16* __restrict__ wpk,
    const float* __restrict__ b0, const float* __restrict__ b1, const float* __restrict__ b2,
    const float* __restrict__ b3, const float* __restrict__ b4,
    float* __restrict__ cam, float* __restrict__ rot, float* __restrict__ trn, float* __restrict__ pc) {
    extern __shared__ char smem[];
    const int bx   = blockIdx.x;            // 2048 = 64 b * 8 mblk * 4 head
    const int b    = bx >> 5;
    const int mblk = (bx >> 2) & 7;
    const int h    = bx & 3;
    const int tid  = threadIdx.x;
    const int wave = tid >> 5;
    const int lane = tid & 31;
    const int g    = lane >> 4;
    const int ml   = lane & 15;

    const int m0   = mblk * 128;
    const int tokA = m0 + wave * 16 + ml;   // token whose cat-row this lane loads
    const int tA = tokA >> 5, sA = tokA & 31;
    const __bf16* pT  = embb + (b * 32 + tA) * 256;
    const __bf16* pS  = embb + (b * 32 + sA) * 256;
    const __bf16* pG1 = g1b + b * 256;
    const __bf16* pG2 = g2b + b * 256;

    char* buf0 = smem;            // 128 KB (act after L0, L2)
    char* buf1 = smem + 131072;   // 64 KB  (act after L1, L3)

    // ---- Layer 0: K=1024 (cat streamed from global bf16), N=512 -> buf0
    {
        const __bf16* wl = wpk + WOFF_L0 + h * 524288 + lane * 16;
        for (int nt = 0; nt < 32; ++nt) {
            v8f c = {};
            const __bf16* wb = wl + nt * 32 * 512;
            __builtin_prefetch(wb + 32 * 512, 0, 0);   // global_prefetch next weight panel
            for (int kt = 0; kt < 32; ++kt) {
                const __bf16* src = (kt < 8) ? pT : (kt < 16) ? pS : (kt < 24) ? pG1 : pG2;
                const int dbase = ((kt & 7) << 5) + (g << 3);
                FragBF a, wf;
                a.u[0]  = *(const uint4*)(src + dbase);
                a.u[1]  = *(const uint4*)(src + dbase + 16);
                wf.u[0] = *(const uint4*)(wb);
                wf.u[1] = *(const uint4*)(wb + 8);
                wb += 512;
                c = __builtin_amdgcn_wmma_f32_16x16x32_bf16(false, a.v, false, wf.v, (short)0, c, false, false);
            }
            const int nabs = nt * 16 + ml;
            const float bias = b0[h * 512 + nabs];
            const int mbase = wave * 16 + (g << 3);
#pragma unroll
            for (int v = 0; v < 8; ++v) {
                float x = c[v] + bias;
                x = x > 0.f ? x : 0.f;
                *(__bf16*)(buf0 + ((mbase + v) * 512 + nabs) * 2) = (__bf16)x;
            }
        }
    }

    // ---- Layers 1..3 (LDS -> LDS, wave-private rows, no barriers needed)
    layer_lds(wpk + WOFF_L1 + h * 131072, b1 + h * 256, buf0, buf1, 16, 16, 256, 512, wave, lane);
    layer_lds(wpk + WOFF_L2 + h * 32768,  b2 + h * 128, buf1, buf0,  8,  8, 128, 256, wave, lane);
    layer_lds(wpk + WOFF_L3 + h * 8192,   b3 + h * 64,  buf0, buf1,  4,  4,  64, 128, wave, lane);

    // ---- Layer 4: K=64, N=4 (padded to 16); head-specific epilogue
    {
        const __bf16* wl = wpk + WOFF_L4 + h * 1024 + lane * 16;
        const char* arow = buf1 + (wave * 16 + ml) * 64 * 2;
        v8f c = {};
        for (int kt = 0; kt < 2; ++kt) {
            FragBF a, wf;
            const int off = (kt * 32 + (g << 3)) * 2;
            a.u[0]  = *(const uint4*)(arow + off);
            a.u[1]  = *(const uint4*)(arow + off + 32);
            wf.u[0] = *(const uint4*)(wl + kt * 512);
            wf.u[1] = *(const uint4*)(wl + kt * 512 + 8);
            c = __builtin_amdgcn_wmma_f32_16x16x32_bf16(false, a.v, false, wf.v, (short)0, c, false, false);
        }
        const int n = ml;                                    // output channel for this lane
        const float bias = (n < 4) ? b4[h * 4 + n] : 0.f;
        const int mtok = b * 1024 + m0 + wave * 16 + (g << 3);
        if (h == 0 && n == 0) {
#pragma unroll
            for (int v = 0; v < 8; ++v) cam[mtok + v] = c[v] + bias;
        } else if (h == 1 && n < 4) {
#pragma unroll
            for (int v = 0; v < 8; ++v) rot[(mtok + v) * 4 + n] = c[v] + bias;
        } else if (h == 2 && n < 3) {
#pragma unroll
            for (int v = 0; v < 8; ++v) trn[(mtok + v) * 4 + n] = c[v] + bias;
        } else if (h == 3 && n == 0) {
#pragma unroll
            for (int v = 0; v < 8; ++v) {
                float x = c[v] + bias;
                pc[mtok + v] = 1.f / (1.f + __expf(-x));
            }
        }
    }
}

// ---------------- finalize: masked reductions (deterministic, no atomics) ----------------
__global__ void __launch_bounds__(256) finalize_kernel(const int* __restrict__ np,
                                                       const float* __restrict__ cam,
                                                       const float* __restrict__ rot,
                                                       const float* __restrict__ trn,
                                                       float* __restrict__ out) {
    __shared__ float red[256];
    int b = blockIdx.x, tid = threadIdx.x;
    int n0 = np[b * 2], n1 = np[b * 2 + 1];
    float acc[8] = {0.f, 0.f, 0.f, 0.f, 0.f, 0.f, 0.f, 0.f};
    for (int idx = tid; idx < 1024; idx += 256) {
        int t = idx >> 5, s = idx & 31;
        bool val = (t < n0) && (s >= n0) && (s < n0 + n1);
        out[66048 + b * 1024 + idx] = val ? 1.f : 0.f;     // valid
        if (val) {
            int p = b * 1024 + idx;
            acc[0] += cam[p];
            acc[1] += rot[p * 4 + 0];
            acc[2] += rot[p * 4 + 1];
            acc[3] += rot[p * 4 + 2];
            acc[4] += rot[p * 4 + 3];
            acc[5] += trn[p * 4 + 0];
            acc[6] += trn[p * 4 + 1];
            acc[7] += trn[p * 4 + 2];
        }
    }
    float pf = (float)(n0 * n1);
    for (int q = 0; q < 8; ++q) {
        red[tid] = acc[q];
        __syncthreads();
        for (int st = 128; st > 0; st >>= 1) {
            if (tid < st) red[tid] += red[tid + st];
            __syncthreads();
        }
        if (tid == 0) {
            float sum = red[0] / pf;
            if (q == 0)      out[b] = 1.f / (1.f + __expf(-sum));          // camera_corr
            else if (q <= 4) out[64 + b * 4 + (q - 1)] = sum;              // rot_res
            else             out[320 + b * 3 + (q - 5)] = sum;             // trans_res
        }
        __syncthreads();
    }
}

// ---------------- host launch ----------------
extern "C" void kernel_launch(void* const* d_in, const int* in_sizes, int n_in,
                              void* d_out, int out_size, void* d_ws, size_t ws_size,
                              hipStream_t stream) {
    (void)in_sizes; (void)n_in; (void)out_size; (void)ws_size;
    const float* emb = (const float*)d_in[0];
    const int*   np  = (const int*)d_in[1];
    const float* w0 = (const float*)d_in[2];
    const float* b0 = (const float*)d_in[3];
    const float* w1 = (const float*)d_in[4];
    const float* b1 = (const float*)d_in[5];
    const float* w2 = (const float*)d_in[6];
    const float* b2 = (const float*)d_in[7];
    const float* w3 = (const float*)d_in[8];
    const float* b3 = (const float*)d_in[9];
    const float* w4 = (const float*)d_in[10];
    const float* b4 = (const float*)d_in[11];

    char* ws = (char*)d_ws;
    __bf16* embb = (__bf16*)(ws + WS_EMBB);
    __bf16* g1b  = (__bf16*)(ws + WS_G1B);
    __bf16* g2b  = (__bf16*)(ws + WS_G2B);
    __bf16* wpk  = (__bf16*)(ws + WS_WPK);
    float*  cam  = (float*)(ws + WS_CAM);
    float*  rot  = (float*)(ws + WS_ROT);
    float*  trn  = (float*)(ws + WS_TRN);
    float*  outp = (float*)d_out;
    float*  pc   = outp + 512;   // plane_corr region in d_out

    conv_emb_kernel<<<2048, 256, 0, stream>>>(emb, embb, 64 * 32 * 256);
    prep_g_kernel<<<64, 256, 0, stream>>>(emb, np, g1b, g2b);

    pack_w_kernel<<<2097152 / 256, 256, 0, stream>>>(w0, wpk + WOFF_L0, 1024, 512, 32, 32, 2097152);
    pack_w_kernel<<< 524288 / 256, 256, 0, stream>>>(w1, wpk + WOFF_L1,  512, 256, 16, 16,  524288);
    pack_w_kernel<<< 131072 / 256, 256, 0, stream>>>(w2, wpk + WOFF_L2,  256, 128,  8,  8,  131072);
    pack_w_kernel<<<  32768 / 256, 256, 0, stream>>>(w3, wpk + WOFF_L3,  128,  64,  4,  4,   32768);
    pack_w_kernel<<<   4096 / 256, 256, 0, stream>>>(w4, wpk + WOFF_L4,   64,   4,  2,  1,    4096);

    mlp_kernel<<<2048, 256, 196608, stream>>>(embb, g1b, g2b, wpk,
                                              b0, b1, b2, b3, b4,
                                              cam, rot, trn, pc);

    finalize_kernel<<<64, 256, 0, stream>>>(np, cam, rot, trn, outp);
}
